// DeepseekMoE_71683004170418
// MI455X (gfx1250) — matrix operand order repaired and checked by reading, not compiled, not run on hardware
//
#include <hip/hip_runtime.h>
#include <hip/hip_bf16.h>
#include <math.h>

// ---- problem constants (match reference) ----
#define T_TOK 512
#define H_DIM 2048
#define E_NUM 32
#define K_TOP 8
#define I_DIM 1408
#define NGRP  8
#define TKG   4
#define CAP   256            // 2*T*K/E
#define RSF   2.5f
#define ISH   (I_DIM * 2)    // shared intermediate = I*NS = 2816
#define BPAD  40             // padded LDS row (halfwords): 80B, 16B-aligned, conflict-free

typedef __attribute__((ext_vector_type(16))) __bf16         v16bf;
typedef __attribute__((ext_vector_type(8)))  float          v8f;
typedef __attribute__((ext_vector_type(16))) unsigned short v16us;
typedef __attribute__((ext_vector_type(8)))  unsigned short v8us;

__device__ __forceinline__ unsigned short f2bf(float f) {
  unsigned int u = __float_as_uint(f);
  return (unsigned short)((u + 0x7FFFu + ((u >> 16) & 1u)) >> 16); // RNE
}

// ---------------- routing: sigmoid + grouped top-k (noaux_tc) ----------------
__global__ void route_kernel(const float* __restrict__ x, const float* __restrict__ gw,
                             const float* __restrict__ bias, int* __restrict__ topk_ids,
                             float* __restrict__ wts) {
  __shared__ float part[8][E_NUM];
  __shared__ float sc[E_NUM];
  __shared__ float sbv[E_NUM];
  int t = blockIdx.x;
  int tid = threadIdx.x;
  int e = tid & 31, c = tid >> 5;           // 8 chunks x 32 experts
  const float* xr = x + (size_t)t * H_DIM;
  float p = 0.f;
  int h0 = c * (H_DIM / 8);
  for (int h = h0; h < h0 + H_DIM / 8; ++h) p += xr[h] * gw[(size_t)h * E_NUM + e];
  part[c][e] = p;
  __syncthreads();
  if (tid < E_NUM) {
    float l = 0.f;
    #pragma unroll
    for (int i = 0; i < 8; ++i) l += part[i][tid];
    float s = 1.f / (1.f + __expf(-l));     // sigmoid (exact-ish; tiny kernel)
    sc[tid]  = s;
    sbv[tid] = s + bias[tid];
  }
  __syncthreads();
  if (tid == 0) {
    float gs[NGRP];
    #pragma unroll
    for (int g = 0; g < NGRP; ++g) {        // sum of top-2 (score+bias) per group
      float m1 = -1e30f, m2 = -1e30f;
      for (int j = 0; j < 4; ++j) {
        float v = sbv[g * 4 + j];
        if (v > m1) { m2 = m1; m1 = v; } else if (v > m2) { m2 = v; }
      }
      gs[g] = m1 + m2;
    }
    bool gsel[NGRP];
    for (int g = 0; g < NGRP; ++g) gsel[g] = false;
    for (int it = 0; it < TKG; ++it) {      // top-4 groups (ties -> lower index)
      float best = -1e30f; int bi = 0;
      for (int g = 0; g < NGRP; ++g) if (!gsel[g] && gs[g] > best) { best = gs[g]; bi = g; }
      gsel[bi] = true;
    }
    bool esel[E_NUM];
    for (int i = 0; i < E_NUM; ++i) esel[i] = false;
    int ids[K_TOP]; float wv[K_TOP]; float wsum = 0.f;
    for (int it = 0; it < K_TOP; ++it) {    // top-8 experts among allowed groups
      float best = -1e30f; int bi = 0;
      for (int q = 0; q < E_NUM; ++q)
        if (gsel[q >> 2] && !esel[q] && sbv[q] > best) { best = sbv[q]; bi = q; }
      esel[bi] = true; ids[it] = bi; wv[it] = sc[bi]; wsum += sc[bi];
    }
    float inv = RSF / wsum;
    for (int it = 0; it < K_TOP; ++it) {
      topk_ids[t * K_TOP + it] = ids[it];
      wts[t * K_TOP + it] = wv[it] * inv;
    }
  }
}

// ---------------- deterministic capacity dispatch (thread per expert) ----------------
__global__ void dispatch_kernel(const int* __restrict__ topk_ids,
                                int* __restrict__ slot_of, int* __restrict__ slot_token) {
  int e = threadIdx.x;                      // 32 threads, one wave
  int cnt = 0;
  for (int i = 0; i < T_TOK * K_TOP; ++i) { // flat order == reference cumsum rank
    if (topk_ids[i] == e) {
      if (cnt < CAP) { slot_of[i] = e * CAP + cnt; slot_token[e * CAP + cnt] = i / K_TOP; }
      else           { slot_of[i] = -1; }
      ++cnt;
    }
  }
  for (int r = cnt; r < CAP; ++r) slot_token[e * CAP + r] = -1;
}

// ---------------- fp32 -> bf16 bulk convert ----------------
__global__ void convert_bf16_kernel(const float* __restrict__ src,
                                    unsigned short* __restrict__ dst, int n4) {
  int i = blockIdx.x * blockDim.x + threadIdx.x;
  if (i < n4) {
    float4 v = ((const float4*)src)[i];
    ushort4 o; o.x = f2bf(v.x); o.y = f2bf(v.y); o.z = f2bf(v.z); o.w = f2bf(v.w);
    ((ushort4*)dst)[i] = o;
  }
}

// ---------------- gather tokens into expert slots (bf16, zero-fill) ----------------
__global__ void gather_kernel(const float* __restrict__ x, const int* __restrict__ slot_token,
                              unsigned short* __restrict__ xe) {
  int b = blockIdx.x;                       // slot index [0, E*CAP)
  int tid = threadIdx.x;
  int tok = slot_token[b];
  unsigned short* dst = xe + (size_t)b * H_DIM;
  if (tok >= 0) {
    const float4* srcv = (const float4*)(x + (size_t)tok * H_DIM);
    for (int i = tid; i < H_DIM / 4; i += 256) {
      float4 v = srcv[i];
      ushort4 o; o.x = f2bf(v.x); o.y = f2bf(v.y); o.z = f2bf(v.z); o.w = f2bf(v.w);
      ((ushort4*)dst)[i] = o;
    }
  } else {
    ushort4 z; z.x = z.y = z.z = z.w = 0;
    for (int i = tid; i < H_DIM / 4; i += 256) ((ushort4*)dst)[i] = z;
  }
}

// ============================================================================
// Block-cooperative WMMA GEMMs.
// Block = 8 waves = 128 output rows x 16 output cols. The fp32 B tile
// (32K x 16N) is staged coalesced -> bf16 -> TRANSPOSED into double-buffered
// LDS; each wave then builds its 32x16 B fragment with 2x ds_load_b128.
// Grid order: mb fastest => blocks sharing an (expert, N-strip) are adjacent,
// so fp32 weights stream from HBM ~once and re-reads hit L2 (192MB).
// ============================================================================

// ---- gate/up GEMM + fused SiLU*up, bf16 out ----
// A: bf16 [batch,M,Kd]; B: fp32 [batch,Kd,2*ldd] (gate cols [0,ldd), up [ldd,2ldd));
// D: bf16 [batch,M,ldd].
__global__ void gemm_gateup_kernel(const unsigned short* __restrict__ A, int lda, long long sA,
                                   const float* __restrict__ B, int ldb, long long sB,
                                   unsigned short* __restrict__ D, int ldd, long long sD,
                                   int Kd, int MB, int NT) {
  __shared__ unsigned short Bt[2][2][16][BPAD];   // [buf][gate/up][n][k(0..31)]
  int tid = threadIdx.x;
  int per = NT * MB;
  int batch = blockIdx.x / per;
  int rem = blockIdx.x - batch * per;
  int nt = rem / MB;
  int mb = rem - nt * MB;
  int w = tid >> 5, lane = tid & 31;
  int l15 = lane & 15, hi = lane >> 4;
  int n0 = nt * 16;

  const unsigned short* Ar = A + (long long)batch * sA
                           + (size_t)((mb * 8 + w) * 16 + l15) * lda;
  int sn = tid & 15;                              // staging column
  int kp = tid >> 4;                              // staging k-pair 0..15
  const float* Bs = B + (long long)batch * sB + n0 + sn;

  v8f accG = {}, accU = {};
  int buf = 0;
  for (int k0 = 0; k0 < Kd; k0 += 32, buf ^= 1) {
    // stage B tile: coalesced fp32 loads, bf16-convert, transpose into LDS
    size_t r0 = (size_t)(k0 + 2 * kp) * ldb;
    size_t r1 = r0 + ldb;
    float g0 = Bs[r0], g1 = Bs[r1];
    float u0 = Bs[r0 + ldd], u1 = Bs[r1 + ldd];
    __builtin_prefetch(Bs + r0 + (size_t)32 * ldb, 0, 1);   // global_prefetch_b8
    *(unsigned int*)&Bt[buf][0][sn][2 * kp] =
        (unsigned int)f2bf(g0) | ((unsigned int)f2bf(g1) << 16);
    *(unsigned int*)&Bt[buf][1][sn][2 * kp] =
        (unsigned int)f2bf(u0) | ((unsigned int)f2bf(u1) << 16);
    // A fragment (16-bit 16x32): lanes<16 -> K{0..15}; lanes>=16 -> K{16..31}
    v8us a0 = *(const v8us*)(Ar + k0 + hi * 16);
    v8us a1 = *(const v8us*)(Ar + k0 + hi * 16 + 8);
    v16us au;
    #pragma unroll
    for (int i = 0; i < 8; ++i) { au[i] = a0[i]; au[i + 8] = a1[i]; }
    __syncthreads();
    // B fragments from LDS (row n = lane&15, K half = hi*16): 2x ds_load_b128 each
    const v8us* pg = (const v8us*)&Bt[buf][0][l15][hi * 16];
    const v8us* pu = (const v8us*)&Bt[buf][1][l15][hi * 16];
    v8us b0 = pg[0], b1 = pg[1];
    v8us c0 = pu[0], c1 = pu[1];
    v16us bg, bu;
    #pragma unroll
    for (int i = 0; i < 8; ++i) {
      bg[i] = b0[i]; bg[i + 8] = b1[i];
      bu[i] = c0[i]; bu[i + 8] = c1[i];
    }
    v16bf av  = __builtin_bit_cast(v16bf, au);
    v16bf bgv = __builtin_bit_cast(v16bf, bg);
    v16bf buv = __builtin_bit_cast(v16bf, bu);
    accG = __builtin_amdgcn_wmma_f32_16x16x32_bf16(false, av, false, bgv, (short)0, accG, false, false);
    accU = __builtin_amdgcn_wmma_f32_16x16x32_bf16(false, av, false, buv, (short)0, accU, false, false);
  }
  // epilogue: silu(gate)*up -> bf16 (v_rcp_f32 instead of IEEE divide)
  unsigned short* Db = D + (long long)batch * sD + n0 + l15;
  int rbase = (mb * 8 + w) * 16 + hi * 8;
  #pragma unroll
  for (int r = 0; r < 8; ++r) {
    float g = accG[r], u = accU[r];
    float m = g * __builtin_amdgcn_rcpf(1.f + __expf(-g)) * u;
    Db[(size_t)(rbase + r) * ldd] = f2bf(m);
  }
}

// ---- down-proj GEMM, fp32 out ----
__global__ void gemm_down_kernel(const unsigned short* __restrict__ A, int lda, long long sA,
                                 const float* __restrict__ B, int ldb, long long sB,
                                 float* __restrict__ D, int ldd, long long sD,
                                 int Kd, int MB, int NT) {
  __shared__ unsigned short Bt[2][16][BPAD];
  int tid = threadIdx.x;
  int per = NT * MB;
  int batch = blockIdx.x / per;
  int rem = blockIdx.x - batch * per;
  int nt = rem / MB;
  int mb = rem - nt * MB;
  int w = tid >> 5, lane = tid & 31;
  int l15 = lane & 15, hi = lane >> 4;
  int n0 = nt * 16;

  const unsigned short* Ar = A + (long long)batch * sA
                           + (size_t)((mb * 8 + w) * 16 + l15) * lda;
  int sn = tid & 15;
  int kp = tid >> 4;
  const float* Bs = B + (long long)batch * sB + n0 + sn;

  v8f acc = {};
  int buf = 0;
  for (int k0 = 0; k0 < Kd; k0 += 32, buf ^= 1) {
    size_t r0 = (size_t)(k0 + 2 * kp) * ldb;
    float g0 = Bs[r0], g1 = Bs[r0 + ldb];
    __builtin_prefetch(Bs + r0 + (size_t)32 * ldb, 0, 1);
    *(unsigned int*)&Bt[buf][sn][2 * kp] =
        (unsigned int)f2bf(g0) | ((unsigned int)f2bf(g1) << 16);
    v8us a0 = *(const v8us*)(Ar + k0 + hi * 16);
    v8us a1 = *(const v8us*)(Ar + k0 + hi * 16 + 8);
    v16us au;
    #pragma unroll
    for (int i = 0; i < 8; ++i) { au[i] = a0[i]; au[i + 8] = a1[i]; }
    __syncthreads();
    const v8us* pb = (const v8us*)&Bt[buf][l15][hi * 16];
    v8us b0 = pb[0], b1 = pb[1];
    v16us bb;
    #pragma unroll
    for (int i = 0; i < 8; ++i) { bb[i] = b0[i]; bb[i + 8] = b1[i]; }
    v16bf av = __builtin_bit_cast(v16bf, au);
    v16bf bv = __builtin_bit_cast(v16bf, bb);
    acc = __builtin_amdgcn_wmma_f32_16x16x32_bf16(false, av, false, bv, (short)0, acc, false, false);
  }
  float* Db = D + (long long)batch * sD + n0 + l15;
  int rbase = (mb * 8 + w) * 16 + hi * 8;
  #pragma unroll
  for (int r = 0; r < 8; ++r) Db[(size_t)(rbase + r) * ldd] = acc[r];
}

// ---------------- combine: out[t] = shared[t] + sum_k w * ye[slot] ----------------
__global__ void combine_kernel(const int* __restrict__ slot_of, const float* __restrict__ wts,
                               const float* __restrict__ ye, float* __restrict__ out) {
  int t = blockIdx.x;
  int tid = threadIdx.x;
  int sl[K_TOP]; float wv[K_TOP];
  #pragma unroll
  for (int k = 0; k < K_TOP; ++k) { sl[k] = slot_of[t * K_TOP + k]; wv[k] = wts[t * K_TOP + k]; }
  float* orow = out + (size_t)t * H_DIM;
  for (int c = tid; c < H_DIM; c += 256) {
    float acc = orow[c];
    #pragma unroll
    for (int k = 0; k < K_TOP; ++k)
      if (sl[k] >= 0) acc += wv[k] * ye[(size_t)sl[k] * H_DIM + c];
    orow[c] = acc;
  }
}

extern "C" void kernel_launch(void* const* d_in, const int* in_sizes, int n_in,
                              void* d_out, int out_size, void* d_ws, size_t ws_size,
                              hipStream_t stream) {
  const float* x    = (const float*)d_in[0];
  // d_in[1] = residual: unused by reference output
  const float* gw   = (const float*)d_in[2];
  const float* bias = (const float*)d_in[3];
  const float* w13  = (const float*)d_in[4];
  const float* w2   = (const float*)d_in[5];
  const float* sgu  = (const float*)d_in[6];
  const float* sdw  = (const float*)d_in[7];
  float* out = (float*)d_out;

  char* ws = (char*)d_ws;
  size_t off = 0;
  auto alloc = [&](size_t bytes) -> char* {
    char* p = ws + off;
    off += (bytes + 255) & ~(size_t)255;
    return p;
  };
  int*   topk_ids = (int*)  alloc((size_t)T_TOK * K_TOP * 4);
  float* wts      = (float*)alloc((size_t)T_TOK * K_TOP * 4);
  int*   slot_of  = (int*)  alloc((size_t)T_TOK * K_TOP * 4);
  int*   slot_tok = (int*)  alloc((size_t)E_NUM * CAP * 4);
  unsigned short* xe   = (unsigned short*)alloc((size_t)E_NUM * CAP * H_DIM * 2);
  unsigned short* xb   = (unsigned short*)alloc((size_t)T_TOK * H_DIM * 2);
  unsigned short* mid  = (unsigned short*)alloc((size_t)E_NUM * CAP * I_DIM * 2);
  unsigned short* mids = (unsigned short*)alloc((size_t)T_TOK * ISH * 2);
  float* ye = (float*)alloc((size_t)E_NUM * CAP * H_DIM * 4);

  // 1) routing
  route_kernel<<<T_TOK, 256, 0, stream>>>(x, gw, bias, topk_ids, wts);
  // 2) deterministic dispatch
  dispatch_kernel<<<1, 32, 0, stream>>>(topk_ids, slot_of, slot_tok);
  // 3) bf16 activations
  {
    int n4 = T_TOK * H_DIM / 4;
    convert_bf16_kernel<<<(n4 + 255) / 256, 256, 0, stream>>>(x, xb, n4);
  }
  gather_kernel<<<E_NUM * CAP, 256, 0, stream>>>(x, slot_tok, xe);

  // 4) routed gate/up + SiLU: 32 x ([256,2048] @ [2048,2816]) -> mid bf16 [32,256,1408]
  {
    int MB = CAP / 128, NT = I_DIM / 16;                      // 2 x 88
    gemm_gateup_kernel<<<E_NUM * NT * MB, 256, 0, stream>>>(
        xe, H_DIM, (long long)CAP * H_DIM,
        w13, 2 * I_DIM, (long long)H_DIM * 2 * I_DIM,
        mid, I_DIM, (long long)CAP * I_DIM,
        H_DIM, MB, NT);
  }
  // 5) shared gate/up + SiLU: [512,2048] @ [2048,5632] -> mids bf16 [512,2816]
  {
    int MB = T_TOK / 128, NT = ISH / 16;                      // 4 x 176
    gemm_gateup_kernel<<<NT * MB, 256, 0, stream>>>(
        xb, H_DIM, 0, sgu, 2 * ISH, 0, mids, ISH, 0, H_DIM, MB, NT);
  }
  // 6) shared down-proj -> d_out (writes every output element)
  {
    int MB = T_TOK / 128, NT = H_DIM / 16;                    // 4 x 128
    gemm_down_kernel<<<NT * MB, 256, 0, stream>>>(
        mids, ISH, 0, sdw, H_DIM, 0, out, H_DIM, 0, ISH, MB, NT);
  }
  // 7) routed down-proj -> ye fp32 [32,256,2048]
  {
    int MB = CAP / 128, NT = H_DIM / 16;                      // 2 x 128
    gemm_down_kernel<<<E_NUM * NT * MB, 256, 0, stream>>>(
        mid, I_DIM, (long long)CAP * I_DIM,
        w2, H_DIM, (long long)I_DIM * H_DIM,
        ye, H_DIM, (long long)CAP * H_DIM,
        I_DIM, MB, NT);
  }
  // 8) weighted gather-combine into d_out
  combine_kernel<<<T_TOK, 256, 0, stream>>>(slot_of, wts, ye, out);
}
